// ParallelRetention_7584912245528
// MI455X (gfx1250) — compile-verified, matrix operand order checked by reading, not run on hardware
//
#include <hip/hip_runtime.h>
#include <hip/hip_bf16.h>

// ---------------------------------------------------------------------------
// ParallelRetention on MI455X (gfx1250): bf16 WMMA pipeline, fp32 accumulate.
// Async global->LDS double-buffered TN GEMM.
//   xf  = pack(x)                 [T, IN]        bf16
//   q   = xf @ Wq^T + bq          [T, INTER]     bf16
//   k   = xf @ Wk^T + bk          [T, INTER]     bf16
//   vT  = (xf @ Wv^T + bv)^T      [INTER, T]     bf16 (stored transposed)
//   S   = gamma ⊙ (q @ k^T)       [T, T]         bf16
//   ret = S @ vT^T                [T, INTER]     bf16
//   out = leaky(ret @ Wo^T + bo)  [T, OUT]       f32  -> d_out
// ---------------------------------------------------------------------------

#define R_DIM     5
#define N_NODE    512
#define T_DIM     4096
#define IN_DIM    2560          // R*N
#define INTER_DIM 4096
#define OUT_DIM   2048
#define LEAKY     0.01f

typedef __attribute__((ext_vector_type(16))) __bf16 v16bf;
typedef __attribute__((ext_vector_type(8)))  __bf16 v8bf;
typedef __attribute__((ext_vector_type(8)))  float  v8f;

__device__ __forceinline__ unsigned short f2bf(float f) {
  union { float f; unsigned u; } v; v.f = f;
  unsigned r = v.u + 0x7FFFu + ((v.u >> 16) & 1u);   // round-to-nearest-even
  return (unsigned short)(r >> 16);
}

// ---- CDNA5 async global->LDS copy (ASYNCcnt-tracked) ----------------------
// VDST = LDS byte address VGPR, VADDR = 64-bit global address (ISA §10.1/§15.18).
__device__ __forceinline__ void async_copy_b128(const unsigned short* g,
                                                const unsigned short* l) {
  unsigned lds = (unsigned)(unsigned long long)(const void*)l;  // low 32b = LDS offset
  asm volatile("global_load_async_to_lds_b128 %0, %1, off"
               :: "v"(lds), "v"(g) : "memory");
}

#if defined(__has_builtin)
#  if __has_builtin(__builtin_amdgcn_s_wait_asynccnt)
#    define WAIT_ASYNC(n) __builtin_amdgcn_s_wait_asynccnt(n)
#  endif
#endif
#ifndef WAIT_ASYNC
#  define WAIT_ASYNC(n) asm volatile("s_wait_asynccnt %0" :: "i"(n) : "memory")
#endif

// --------------------------- pack / convert kernels ------------------------

// xf[t, n*R + r] = bf16( x[r, n, t] )
__global__ __launch_bounds__(256) void pack_x_kernel(const float* __restrict__ x,
                                                     unsigned short* __restrict__ xf) {
  size_t i = (size_t)blockIdx.x * 256 + threadIdx.x;   // over T*IN
  if (i >= (size_t)T_DIM * IN_DIM) return;
  int t = (int)(i / IN_DIM);
  int c = (int)(i % IN_DIM);
  int n = c / R_DIM;
  int r = c % R_DIM;
  float v = x[((size_t)r * N_NODE + n) * T_DIM + t];
  xf[i] = f2bf(v);
}

__global__ __launch_bounds__(256) void cvt_bf16_kernel(const float* __restrict__ in,
                                                       unsigned short* __restrict__ out,
                                                       size_t n) {
  size_t i = (size_t)blockIdx.x * 256 + threadIdx.x;
  if (i < n) out[i] = f2bf(in[i]);
}

// ------------------------------- GEMM kernel -------------------------------
// C = epi( A[M,K] @ B[N,K]^T + bias ),  A/B bf16 row-major, K%32==0, M%128==0,
// N%128==0.  256 threads = 8 waves (2x4); each wave owns a 64x32 subtile =
// 4x2 v_wmma_f32_16x16x32_bf16 accumulators.  LDS tiles double-buffered and
// filled with global_load_async_to_lds_b128 (ASYNCcnt).

#define BM 128
#define BN 128
#define BK 32
#define ASYNC_PER_THREAD 4     // async b128 copies issued per thread per stage

#define EPI_NONE  0
#define EPI_GAMMA 1
#define ST_BF16   0
#define ST_BF16_T 1
#define ST_F32_LK 2

template <int EPI, int STORE>
__global__ __launch_bounds__(256) void gemm_tn_bf16(
    const unsigned short* __restrict__ A, int lda,
    const unsigned short* __restrict__ B, int ldb,
    const float* __restrict__ bias,          // nullable
    const float* __restrict__ gamma, int ldg,
    void* __restrict__ Cptr, int ldc,
    int Kk) {
  __shared__ __align__(16) unsigned short sA[2][BM * BK];   // 2 x 8 KB
  __shared__ __align__(16) unsigned short sB[2][BN * BK];   // 2 x 8 KB

  const int tid   = threadIdx.x;
  const int lane  = tid & 31;
  const int wave  = tid >> 5;
  const int waveM = wave >> 2;                 // 0..1  (rows of 64)
  const int waveN = wave & 3;                  // 0..3  (cols of 32)
  const int blockM = blockIdx.y * BM;
  const int blockN = blockIdx.x * BN;

  // Per-thread global->LDS staging coordinates (16B chunks; 512 chunks/tile).
  const int aR0 = tid >> 2,          cC = (tid & 3) * 8;
  const int aR1 = (tid + 256) >> 2;

  const unsigned short* gA0 = A + (size_t)(blockM + aR0) * lda + cC;
  const unsigned short* gA1 = A + (size_t)(blockM + aR1) * lda + cC;
  const unsigned short* gB0 = B + (size_t)(blockN + aR0) * ldb + cC;
  const unsigned short* gB1 = B + (size_t)(blockN + aR1) * ldb + cC;

  auto stage = [&](int k0, int buf) {
    async_copy_b128(gA0 + k0, &sA[buf][aR0 * BK + cC]);
    async_copy_b128(gA1 + k0, &sA[buf][aR1 * BK + cC]);
    async_copy_b128(gB0 + k0, &sB[buf][aR0 * BK + cC]);
    async_copy_b128(gB1 + k0, &sB[buf][aR1 * BK + cC]);
  };

  // ISA 7.12.2: 16-bit A/B fragment lane mapping.
  const int frow  = lane & 15;
  const int kbase = (lane >> 4) * 8;           // lanes 16-31 -> K+8

  v8f acc[4][2];
#pragma unroll
  for (int i = 0; i < 4; ++i)
#pragma unroll
    for (int j = 0; j < 2; ++j) acc[i][j] = (v8f){};

  const int nsteps = Kk / BK;
  stage(0, 0);

  for (int s = 0; s < nsteps; ++s) {
    const int buf = s & 1;
    const bool pipe = (s + 1 < nsteps);
    if (pipe) {
      stage((s + 1) * BK, buf ^ 1);
      WAIT_ASYNC(ASYNC_PER_THREAD);            // stage s complete, s+1 in flight
    } else {
      WAIT_ASYNC(0);
    }
    __syncthreads();

    // ---- fragments from LDS ----------------------------------------------
    v16bf af[4], bf[2];
#pragma unroll
    for (int i = 0; i < 4; ++i) {
      const __bf16* p =
          (const __bf16*)(&sA[buf][(waveM * 64 + i * 16 + frow) * BK + kbase]);
      v8bf lo = *(const v8bf*)p, hi = *(const v8bf*)(p + 16);
#pragma unroll
      for (int h = 0; h < 8; ++h) { af[i][h] = lo[h]; af[i][h + 8] = hi[h]; }
    }
#pragma unroll
    for (int j = 0; j < 2; ++j) {
      const __bf16* p =
          (const __bf16*)(&sB[buf][(waveN * 32 + j * 16 + frow) * BK + kbase]);
      v8bf lo = *(const v8bf*)p, hi = *(const v8bf*)(p + 16);
#pragma unroll
      for (int h = 0; h < 8; ++h) { bf[j][h] = lo[h]; bf[j][h + 8] = hi[h]; }
    }

    // ---- 4x2 WMMA ---------------------------------------------------------
#pragma unroll
    for (int i = 0; i < 4; ++i)
#pragma unroll
      for (int j = 0; j < 2; ++j)
        acc[i][j] = __builtin_amdgcn_wmma_f32_16x16x32_bf16(
            false, af[i], false, bf[j], (short)0, acc[i][j], false, false);

    __syncthreads();                           // buf reusable for stage s+2
  }

  // ---- epilogue + store (C/D layout: VGPR r -> M=r (+8 for lanes 16-31),
  //      lane&15 -> N) -------------------------------------------------------
  const int mhi = (lane >> 4) * 8;
  const int nlo = lane & 15;
#pragma unroll
  for (int i = 0; i < 4; ++i) {
#pragma unroll
    for (int j = 0; j < 2; ++j) {
      const int rbase = blockM + waveM * 64 + i * 16 + mhi;
      const int col   = blockN + waveN * 32 + j * 16 + nlo;
#pragma unroll
      for (int r = 0; r < 8; ++r) {
        const int row = rbase + r;
        float x = acc[i][j][r];
        if (bias) x += bias[col];
        if (EPI == EPI_GAMMA) x *= gamma[(size_t)row * ldg + col];
        if (STORE == ST_F32_LK) {
          x = (x > 0.f) ? x : LEAKY * x;
          ((float*)Cptr)[(size_t)row * ldc + col] = x;
        } else if (STORE == ST_BF16_T) {
          ((unsigned short*)Cptr)[(size_t)col * ldc + row] = f2bf(x);
        } else {
          ((unsigned short*)Cptr)[(size_t)row * ldc + col] = f2bf(x);
        }
      }
    }
  }
}

// ------------------------------ launch ------------------------------------

// Workspace layout (bytes, all offsets 256-aligned); aliasing reuses dead bufs.
#define SZ_XF  ((size_t)T_DIM * IN_DIM * 2)        // 20,971,520
#define SZ_WQ  ((size_t)INTER_DIM * IN_DIM * 2)    // 20,971,520
#define SZ_WO  ((size_t)OUT_DIM * INTER_DIM * 2)   // 16,777,216
#define SZ_TT  ((size_t)T_DIM * INTER_DIM * 2)     // 33,554,432

#define OFF_XF  ((size_t)0)
#define OFF_WQ  (OFF_XF + SZ_XF)
#define OFF_WK  (OFF_WQ + SZ_WQ)
#define OFF_WV  (OFF_WK + SZ_WQ)
#define OFF_WO  (OFF_WV + SZ_WQ)
#define OFF_Q   (OFF_WO + SZ_WO)
#define OFF_K   (OFF_Q  + SZ_TT)
#define OFF_VT  (OFF_K  + SZ_TT)
#define OFF_S   (OFF_WQ)   // alias: Wq+Wk region dead once q,k are computed
#define OFF_RET (OFF_Q)    // alias: q dead once scores are computed

extern "C" void kernel_launch(void* const* d_in, const int* in_sizes, int n_in,
                              void* d_out, int out_size, void* d_ws, size_t ws_size,
                              hipStream_t stream) {
  (void)in_sizes; (void)n_in; (void)out_size; (void)ws_size;
  const float* x  = (const float*)d_in[0];
  const float* dg = (const float*)d_in[1];
  const float* Wq = (const float*)d_in[2];
  const float* bq = (const float*)d_in[3];
  const float* Wk = (const float*)d_in[4];
  const float* bk = (const float*)d_in[5];
  const float* Wv = (const float*)d_in[6];
  const float* bv = (const float*)d_in[7];
  const float* Wo = (const float*)d_in[8];
  const float* bo = (const float*)d_in[9];
  float* out = (float*)d_out;

  char* ws = (char*)d_ws;
  unsigned short* XF  = (unsigned short*)(ws + OFF_XF);
  unsigned short* WQb = (unsigned short*)(ws + OFF_WQ);
  unsigned short* WKb = (unsigned short*)(ws + OFF_WK);
  unsigned short* WVb = (unsigned short*)(ws + OFF_WV);
  unsigned short* WOb = (unsigned short*)(ws + OFF_WO);
  unsigned short* Qb  = (unsigned short*)(ws + OFF_Q);
  unsigned short* Kb  = (unsigned short*)(ws + OFF_K);
  unsigned short* VTb = (unsigned short*)(ws + OFF_VT);
  unsigned short* Sb  = (unsigned short*)(ws + OFF_S);
  unsigned short* Rb  = (unsigned short*)(ws + OFF_RET);

  // ---- pack + precision conversion ----------------------------------------
  {
    size_t n = (size_t)T_DIM * IN_DIM;
    pack_x_kernel<<<(unsigned)((n + 255) / 256), 256, 0, stream>>>(x, XF);
    size_t nw = (size_t)INTER_DIM * IN_DIM;
    cvt_bf16_kernel<<<(unsigned)((nw + 255) / 256), 256, 0, stream>>>(Wq, WQb, nw);
    cvt_bf16_kernel<<<(unsigned)((nw + 255) / 256), 256, 0, stream>>>(Wk, WKb, nw);
    cvt_bf16_kernel<<<(unsigned)((nw + 255) / 256), 256, 0, stream>>>(Wv, WVb, nw);
    size_t no = (size_t)OUT_DIM * INTER_DIM;
    cvt_bf16_kernel<<<(unsigned)((no + 255) / 256), 256, 0, stream>>>(Wo, WOb, no);
  }

  dim3 blk(256);
  // ---- q, k, vT ------------------------------------------------------------
  {
    dim3 grid(INTER_DIM / BN, T_DIM / BM);
    gemm_tn_bf16<EPI_NONE, ST_BF16><<<grid, blk, 0, stream>>>(
        XF, IN_DIM, WQb, IN_DIM, bq, nullptr, 0, Qb, INTER_DIM, IN_DIM);
    gemm_tn_bf16<EPI_NONE, ST_BF16><<<grid, blk, 0, stream>>>(
        XF, IN_DIM, WKb, IN_DIM, bk, nullptr, 0, Kb, INTER_DIM, IN_DIM);
    gemm_tn_bf16<EPI_NONE, ST_BF16_T><<<grid, blk, 0, stream>>>(
        XF, IN_DIM, WVb, IN_DIM, bv, nullptr, 0, VTb, T_DIM, IN_DIM);
  }
  // ---- S = gamma ⊙ (q @ k^T) ----------------------------------------------
  {
    dim3 grid(T_DIM / BN, T_DIM / BM);
    gemm_tn_bf16<EPI_GAMMA, ST_BF16><<<grid, blk, 0, stream>>>(
        Qb, INTER_DIM, Kb, INTER_DIM, nullptr, dg, T_DIM, Sb, T_DIM, INTER_DIM);
  }
  // ---- ret = S @ vT^T ------------------------------------------------------
  {
    dim3 grid(INTER_DIM / BN, T_DIM / BM);
    gemm_tn_bf16<EPI_NONE, ST_BF16><<<grid, blk, 0, stream>>>(
        Sb, T_DIM, VTb, T_DIM, nullptr, nullptr, 0, Rb, INTER_DIM, T_DIM);
  }
  // ---- out = leaky(ret @ Wo^T + bo) ---------------------------------------
  {
    dim3 grid(OUT_DIM / BN, T_DIM / BM);
    gemm_tn_bf16<EPI_NONE, ST_F32_LK><<<grid, blk, 0, stream>>>(
        Rb, INTER_DIM, WOb, INTER_DIM, bo, nullptr, 0, out, OUT_DIM, INTER_DIM);
  }
}
